// AutoCorrelation_spa_tem_80951543595624
// MI455X (gfx1250) — matrix operand order repaired and checked by reading, not compiled, not run on hardware
//
#include <hip/hip_runtime.h>

#define Bn   32
#define Ln   192
#define Hn   8
#define En   64
#define HEn  512
#define TOPK 5
#define QSTR (Ln*HEn)   // 98304 floats per batch of Q/K/V
#define GSTR (Ln*Ln)    // 36864 floats per batch of G2/M

typedef __attribute__((ext_vector_type(16))) __bf16 v16bf;
typedef __attribute__((ext_vector_type(8)))  float  v8f;

// ---- fragment helpers: native f32->bf16 (RNE) via v_cvt_pk_bf16_f32 ---------
__device__ __forceinline__ void pack4(v16bf& v, int o, float4 f) {
  v[o + 0] = (__bf16)f.x; v[o + 1] = (__bf16)f.y;
  v[o + 2] = (__bf16)f.z; v[o + 3] = (__bf16)f.w;
}

// A-operand (16x32 bf16): lane m = lane&15, k-base akb = (lane>>4)*8;
// element p maps to k = akb + p (+8 if p>=8)  -> two contiguous float4 pairs.
__device__ __forceinline__ v16bf load_afrag(const float* base /* row + ks + akb */) {
  v16bf v;
  const float4* p0 = (const float4*)base;
  const float4* p1 = (const float4*)(base + 16);
  pack4(v, 0, p0[0]); pack4(v, 4,  p0[1]);
  pack4(v, 8, p1[0]); pack4(v, 12, p1[1]);
  return v;
}

// B-operand (32x16 bf16): lane n = lane&15, k-base bkb = (lane>>4)*8;
// element p maps to k = bkb + p (+8 if p>=8)  -> two contiguous float4 pairs.
__device__ __forceinline__ v16bf load_bfrag(const float* base /* row + ks + bkb */) {
  v16bf v;
  const float4* p = (const float4*)base;
  const float4* q = (const float4*)(base + 16);
  pack4(v, 0, p[0]); pack4(v, 4, p[1]); pack4(v, 8, q[0]); pack4(v, 12, q[1]);
  return v;
}

#define WMMA_BF16(a, b, c) \
  __builtin_amdgcn_wmma_f32_16x16x32_bf16(false, (a), false, (b), (short)0, (c), false, false)

// ---------------------------------------------------------------- init
__global__ void k_init(float* __restrict__ ws_mean) {
  int i = blockIdx.x * blockDim.x + threadIdx.x;
  if (i < Bn * Ln) ws_mean[i] = 0.f;
}

// ---------------------------------------------------------------- Gram + diagonal fold
// G[b][t1,t2] = sum_c Q[b,t1,c]*K[b,t2,c]; element folds into tau=(t1-t2)%L.
// 4-wave blocks; each wave: one 16x64 strip (A fragment reused across 4 B tiles).
// jobs per batch = 12 ti * 3 groups = 36 (divisible by 4 -> block never straddles b).
__global__ void __launch_bounds__(128) k_corr(const float* __restrict__ Q,
                                              const float* __restrict__ K,
                                              float* __restrict__ ws_mean) {
  __shared__ float smean[Ln];
  const int lane = threadIdx.x & 31;
  const int wave = threadIdx.x >> 5;
  const int job  = blockIdx.x * 4 + wave;
  const int b    = job / 36;
  const int rj   = job % 36;
  const int ti   = rj / 3;
  const int tj0  = (rj % 3) * 4;

  for (int i = threadIdx.x; i < Ln; i += 128) smean[i] = 0.f;
  __syncthreads();

  const int m   = lane & 15;
  const int akb = (lane >> 4) * 8;
  const int bkb = (lane >> 4) * 8;
  const float* arow = Q + (size_t)b * QSTR + (size_t)(ti * 16 + m) * HEn;
  const float* br0  = K + (size_t)b * QSTR + (size_t)((tj0 + 0) * 16 + m) * HEn;
  const float* br1  = K + (size_t)b * QSTR + (size_t)((tj0 + 1) * 16 + m) * HEn;
  const float* br2  = K + (size_t)b * QSTR + (size_t)((tj0 + 2) * 16 + m) * HEn;
  const float* br3  = K + (size_t)b * QSTR + (size_t)((tj0 + 3) * 16 + m) * HEn;

  v8f acc[4] = {};
  for (int ks = 0; ks < HEn; ks += 32) {
    v16bf a = load_afrag(arow + ks + akb);
    acc[0] = WMMA_BF16(a, load_bfrag(br0 + ks + bkb), acc[0]);
    acc[1] = WMMA_BF16(a, load_bfrag(br1 + ks + bkb), acc[1]);
    acc[2] = WMMA_BF16(a, load_bfrag(br2 + ks + bkb), acc[2]);
    acc[3] = WMMA_BF16(a, load_bfrag(br3 + ks + bkb), acc[3]);
  }

  const int msel = (lane >> 4) * 8;
#pragma unroll
  for (int c = 0; c < 4; ++c) {
#pragma unroll
    for (int r = 0; r < 8; ++r) {
      int t1  = ti * 16 + r + msel;
      int t2  = (tj0 + c) * 16 + m;
      int tau = t1 - t2; if (tau < 0) tau += Ln;
      atomicAdd(&smean[tau], acc[c][r]);
    }
  }
  __syncthreads();
  for (int i = threadIdx.x; i < Ln; i += 128)
    atomicAdd(&ws_mean[b * Ln + i], smean[i]);
}

// ---------------------------------------------------------------- top-5 + softmax weights
__global__ void k_topk(const float* __restrict__ ws_mean,
                       float* __restrict__ ws_w, int* __restrict__ ws_idx) {
  __shared__ float g[Ln];
  __shared__ int   sidx[TOPK];
  const int t = threadIdx.x;
  for (int i = t; i < Ln; i += blockDim.x) {
    float s = 0.f;
    for (int b = 0; b < Bn; ++b) s += ws_mean[b * Ln + i];
    g[i] = s;
  }
  __syncthreads();
  if (t == 0) {
    for (int i = 0; i < TOPK; ++i) {
      int best = 0; float bv = -3.4e38f;
      for (int j = 0; j < Ln; ++j) {
        bool taken = false;
        for (int p = 0; p < i; ++p) taken |= (sidx[p] == j);
        if (!taken && g[j] > bv) { bv = g[j]; best = j; }
      }
      sidx[i] = best; ws_idx[i] = best;
    }
  }
  __syncthreads();
  if (t < Bn) {
    float w[TOPK]; float mx = -3.4e38f;
    for (int i = 0; i < TOPK; ++i) {
      w[i] = ws_mean[t * Ln + sidx[i]] * (1.f / (Hn * En));
      mx = fmaxf(mx, w[i]);
    }
    float s = 0.f;
    for (int i = 0; i < TOPK; ++i) { w[i] = __expf(w[i] - mx); s += w[i]; }
    for (int i = 0; i < TOPK; ++i) ws_w[t * TOPK + i] = w[i] / s;
  }
}

// ---------------------------------------------------------------- G2 = k_spa^T @ v_spa
// G2[b][t,l] = sum_n keys_flat[b][n*L+t] * values_flat[b][l*HE+n]
// Strided A gather reused across 4 contiguous B tiles; coalesced stores.
__global__ void __launch_bounds__(128) k_g2(const float* __restrict__ K,
                                            const float* __restrict__ V,
                                            float* __restrict__ ws_g2) {
  const int lane = threadIdx.x & 31;
  const int wave = threadIdx.x >> 5;
  const int job  = blockIdx.x * 4 + wave;
  const int b    = job / 36;
  const int rj   = job % 36;
  const int ti   = rj / 3;
  const int tj0  = (rj % 3) * 4;
  const int m    = lane & 15;
  const int akb  = (lane >> 4) * 8;
  const int bkb  = (lane >> 4) * 8;
  const float* kb = K + (size_t)b * QSTR;
  const float* vr0 = V + (size_t)b * QSTR + (size_t)((tj0 + 0) * 16 + m) * HEn;
  const float* vr1 = V + (size_t)b * QSTR + (size_t)((tj0 + 1) * 16 + m) * HEn;
  const float* vr2 = V + (size_t)b * QSTR + (size_t)((tj0 + 2) * 16 + m) * HEn;
  const float* vr3 = V + (size_t)b * QSTR + (size_t)((tj0 + 3) * 16 + m) * HEn;
  const int t1 = ti * 16 + m;

  v8f acc[4] = {};
  for (int ks = 0; ks < HEn; ks += 32) {
    v16bf a;
#pragma unroll
    for (int p = 0; p < 16; ++p) {
      int kk = ks + akb + p + ((p >= 8) ? 8 : 0);    // A 16x32 k-map
      a[p] = (__bf16)kb[(size_t)kk * Ln + t1];       // stride-L gather (reused 4x)
    }
    acc[0] = WMMA_BF16(a, load_bfrag(vr0 + ks + bkb), acc[0]);
    acc[1] = WMMA_BF16(a, load_bfrag(vr1 + ks + bkb), acc[1]);
    acc[2] = WMMA_BF16(a, load_bfrag(vr2 + ks + bkb), acc[2]);
    acc[3] = WMMA_BF16(a, load_bfrag(vr3 + ks + bkb), acc[3]);
  }

  const int msel = (lane >> 4) * 8;
#pragma unroll
  for (int c = 0; c < 4; ++c) {
#pragma unroll
    for (int r = 0; r < 8; ++r) {
      int tt = ti * 16 + r + msel;
      int ll = (tj0 + c) * 16 + m;
      ws_g2[(size_t)b * GSTR + (size_t)tt * Ln + ll] = acc[c][r];
    }
  }
}

// ---------------------------------------------------------------- M = sum_i w_i * doubly-rolled G2
__global__ void k_combine(const float* __restrict__ ws_g2, const float* __restrict__ ws_w,
                          const int* __restrict__ ws_idx, float* __restrict__ ws_M) {
  int gid = blockIdx.x * blockDim.x + threadIdx.x;
  if (gid >= Bn * GSTR) return;
  int b  = gid / GSTR;
  int r  = gid % GSTR;
  int t1 = r / Ln, t2 = r % Ln;
  float s = 0.f;
#pragma unroll
  for (int i = 0; i < TOPK; ++i) {
    int d = ws_idx[i];
    int a = t1 + d; if (a >= Ln) a -= Ln;
    int c = t2 + d; if (c >= Ln) c -= Ln;
    s += ws_w[b * TOPK + i] * ws_g2[(size_t)b * GSTR + (size_t)a * Ln + c];
  }
  ws_M[gid] = s;
}

// ---------------------------------------------------------------- transposed final GEMM
// out2d^T[t,m] = sum_{t'} M[t',t] * q_spa[b,m,t']   (t = M-dim rows, m = N-dim lanes)
// A = M^T (stride-L gather, reused across 4 B tiles), B = q_spa^T (contiguous b128),
// stores: out[b, t, m..m+15] -> fully coalesced 64B runs.
// jobs per batch = 12 ti * 8 col-groups = 96 (divisible by 4).
__global__ void __launch_bounds__(128) k_final(const float* __restrict__ Q,
                                               const float* __restrict__ wsM,
                                               float* __restrict__ out) {
  const int lane = threadIdx.x & 31;
  const int wave = threadIdx.x >> 5;
  const int job  = blockIdx.x * 4 + wave;
  const int b    = job / 96;
  const int rj   = job % 96;
  const int ti   = rj / 8;          // row tile over t (0..11)
  const int mj0  = (rj % 8) * 4;    // col tile group over m (0..31 in steps of 4)
  const int m    = lane & 15;
  const int akb  = (lane >> 4) * 8;
  const int bkb  = (lane >> 4) * 8;
  const float* Mb  = wsM + (size_t)b * GSTR;
  const float* qb  = Q + (size_t)b * QSTR;
  const float* qr0 = qb + (size_t)((mj0 + 0) * 16 + m) * Ln;  // q_spa rows (contig in t')
  const float* qr1 = qb + (size_t)((mj0 + 1) * 16 + m) * Ln;
  const float* qr2 = qb + (size_t)((mj0 + 2) * 16 + m) * Ln;
  const float* qr3 = qb + (size_t)((mj0 + 3) * 16 + m) * Ln;
  const int trow = ti * 16 + m;

  v8f acc[4] = {};
  for (int ks = 0; ks < Ln; ks += 32) {
    v16bf a;
#pragma unroll
    for (int p = 0; p < 16; ++p) {
      int kk = ks + akb + p + ((p >= 8) ? 8 : 0);     // A 16x32 k-map
      a[p] = (__bf16)Mb[(size_t)kk * Ln + trow];      // M^T gather (reused 4x)
    }
    acc[0] = WMMA_BF16(a, load_bfrag(qr0 + ks + bkb), acc[0]);
    acc[1] = WMMA_BF16(a, load_bfrag(qr1 + ks + bkb), acc[1]);
    acc[2] = WMMA_BF16(a, load_bfrag(qr2 + ks + bkb), acc[2]);
    acc[3] = WMMA_BF16(a, load_bfrag(qr3 + ks + bkb), acc[3]);
  }

  const int msel = (lane >> 4) * 8;
#pragma unroll
  for (int c = 0; c < 4; ++c) {
#pragma unroll
    for (int r = 0; r < 8; ++r) {
      int tt = ti * 16 + r + msel;                    // output time index
      int mm = (mj0 + c) * 16 + m;                    // output (h,e) flat index
      out[(size_t)b * QSTR + (size_t)tt * HEn + mm] = acc[c][r];  // V[b,t,h,e]
    }
  }
}

extern "C" void kernel_launch(void* const* d_in, const int* in_sizes, int n_in,
                              void* d_out, int out_size, void* d_ws, size_t ws_size,
                              hipStream_t stream) {
  const float* q = (const float*)d_in[0];
  const float* k = (const float*)d_in[1];
  const float* v = (const float*)d_in[2];
  // d_in[3] (adj) and d_in[4] (attn_mask) are unused by the reference.

  float* ws      = (float*)d_ws;
  float* ws_mean = ws;                              // Bn*Ln  = 6144
  float* ws_w    = ws + 6144;                       // Bn*5   = 160
  int*   ws_idx  = (int*)(ws + 6336);               // 5 ints (64-slot pad)
  float* ws_g2   = ws + 6400;                       // Bn*GSTR = 1179648
  float* ws_M    = ws_g2 + (size_t)Bn * GSTR;       // Bn*GSTR = 1179648
  float* out     = (float*)d_out;

  k_init   <<<dim3((Bn * Ln + 255) / 256), dim3(256), 0, stream>>>(ws_mean);
  k_corr   <<<dim3(Bn * 36 / 4),           dim3(128), 0, stream>>>(q, k, ws_mean);
  k_topk   <<<dim3(1),                     dim3(64),  0, stream>>>(ws_mean, ws_w, ws_idx);
  k_g2     <<<dim3(Bn * 36 / 4),           dim3(128), 0, stream>>>(k, v, ws_g2);
  k_combine<<<dim3((Bn * GSTR + 255)/256), dim3(256), 0, stream>>>(ws_g2, ws_w, ws_idx, ws_M);
  k_final  <<<dim3(Bn * 96 / 4),           dim3(128), 0, stream>>>(q, ws_M, out);
}
